// LiquidNeuralODE_47605417508988
// MI455X (gfx1250) — matrix operand order repaired and checked by reading, not compiled, not run on hardware
//
#include <hip/hip_runtime.h>
#include <hip/hip_bf16.h>

typedef __attribute__((ext_vector_type(16))) _Float16 v16h;
typedef __attribute__((ext_vector_type(8)))  _Float16 v8h;
typedef __attribute__((ext_vector_type(8)))  float    v8f;

union V16 { v16h v; v8h h[2]; };

// Epilogue modes (compile-time -> branch-free epilogues)
//  0: +bias -> outF
//  1: +bias, tanh -> outH
//  2: +bias, tanh -> outF + outH
//  3: (no bias) -> outF
//  4: +resid -> outF + outH
//  5: gate: s=sigmoid(acc+bias); out = s*t + (1-s)*h -> outF + outH
// =====================================================================
// GEMM:  out[M,N] = A[M,K](f16,row-major) * W[N,K](f16,row-major)^T
// Block = 128 threads = 4 waves, block tile 64(M) x 128(N),
// wave tile 32x64 = 2x4 of v_wmma_f32_16x16x32_f16.
// Fragments are double-buffered (unroll-by-2 ping-pong) so the k+32 loads
// overlap the 8 WMMAs of slab k.  Requires K % 64 == 0 (true for all dims).
// =====================================================================
template<int MODE>
__global__ __launch_bounds__(128)
void wmma_gemm_kernel(const _Float16* __restrict__ A,
                      const _Float16* __restrict__ W,
                      const float* __restrict__ bias,
                      const float* __restrict__ resid,
                      const float* __restrict__ tbuf,
                      const float* __restrict__ hbuf,
                      float* __restrict__ outF,
                      _Float16* __restrict__ outH,
                      int M, int N, int K)
{
    const int wid  = (int)(threadIdx.x >> 5);
    const int lane = (int)(threadIdx.x & 31);
    const int g    = lane >> 4;     // half-wave select
    const int ln16 = lane & 15;

    const int waveM = (int)blockIdx.y * 64  + (wid >> 1) * 32;
    const int waveN = (int)blockIdx.x * 128 + (wid & 1) * 64;

    // Per-fragment row base pointers (advance by k inside the loop).
    const _Float16* apr[2];
#pragma unroll
    for (int i = 0; i < 2; ++i)
        apr[i] = A + (size_t)(waveM + i*16 + ln16) * K + g*8;
    const _Float16* bpr[4];
#pragma unroll
    for (int j = 0; j < 4; ++j)
        bpr[j] = W + (size_t)(waveN + j*16 + ln16) * K + g*8;

    const v8f zero = {0.f,0.f,0.f,0.f,0.f,0.f,0.f,0.f};
    v8f acc[2][4];
#pragma unroll
    for (int i = 0; i < 2; ++i)
#pragma unroll
        for (int j = 0; j < 4; ++j) acc[i][j] = zero;

    auto loadFrags = [&](V16 (&a)[2], V16 (&b)[4], int koff) {
#pragma unroll
        for (int i = 0; i < 2; ++i) {
            a[i].h[0] = *(const v8h*)(apr[i] + koff);
            a[i].h[1] = *(const v8h*)(apr[i] + koff + 16);
        }
#pragma unroll
        for (int j = 0; j < 4; ++j) {
            b[j].h[0] = *(const v8h*)(bpr[j] + koff);
            b[j].h[1] = *(const v8h*)(bpr[j] + koff + 16);
        }
    };
    auto mma = [&](V16 (&a)[2], V16 (&b)[4]) {
#pragma unroll
        for (int j = 0; j < 4; ++j)
#pragma unroll
            for (int i = 0; i < 2; ++i)
                acc[i][j] = __builtin_amdgcn_wmma_f32_16x16x32_f16(
                    false, a[i].v, false, b[j].v, (short)0, acc[i][j], false, false);
    };

    V16 a0[2], b0[4], a1[2], b1[4];
    loadFrags(a0, b0, 0);
    for (int k0 = 0; k0 < K; k0 += 64) {
        loadFrags(a1, b1, k0 + 32);          // in flight while slab k0 computes
        mma(a0, b0);
        if (k0 + 64 < K) loadFrags(a0, b0, k0 + 64);
        mma(a1, b1);
    }

    constexpr bool HAS_BIAS  = (MODE == 0 || MODE == 1 || MODE == 2 || MODE == 5);
    constexpr bool HAS_RESID = (MODE == 4);
    constexpr bool HAS_TANH  = (MODE == 1 || MODE == 2);
    constexpr bool IS_GATE   = (MODE == 5);
    constexpr bool HAS_F     = (MODE == 0 || MODE == 2 || MODE == 3 || MODE == 4 || MODE == 5);
    constexpr bool HAS_H     = (MODE == 1 || MODE == 2 || MODE == 4 || MODE == 5);

    // C layout: VGPR r of tile (i,j) -> row waveM+16i+8g+r, col waveN+16j+ln16
#pragma unroll
    for (int i = 0; i < 2; ++i) {
#pragma unroll
        for (int j = 0; j < 4; ++j) {
            const int col = waveN + j*16 + ln16;
            float bv = 0.f;
            if (HAS_BIAS) bv = bias[col];
#pragma unroll
            for (int r = 0; r < 8; ++r) {
                const int row = waveM + i*16 + g*8 + r;
                const size_t idx = (size_t)row * N + col;
                float val = acc[i][j][r] + bv;
                if (HAS_RESID) val += resid[idx];
                if (HAS_TANH)  val = tanhf(val);
                if (IS_GATE) {
                    const float s = 1.f / (1.f + __expf(-val));
                    val = s * tbuf[idx] + (1.f - s) * hbuf[idx];
                }
                if (HAS_F) outF[idx] = val;
                if (HAS_H) outH[idx] = (_Float16)val;
            }
        }
    }
}

// ---------------- f32 -> f16 convert ----------------
__global__ void cvt_kernel(const float* __restrict__ s, _Float16* __restrict__ d, int n)
{
    int i = (int)(blockIdx.x * blockDim.x + threadIdx.x);
    if (i < n) d[i] = (_Float16)s[i];
}

// ---------------- LayerNorm (+optional tanh-gelu) ----------------
__global__ __launch_bounds__(256)
void ln_kernel(const float* __restrict__ in, const float* __restrict__ gam,
               const float* __restrict__ bet, float* __restrict__ outF,
               _Float16* __restrict__ outH, int N, int act)
{
    const int row  = (int)blockIdx.x;
    const int tid  = (int)threadIdx.x;
    const int wid  = tid >> 5, lane = tid & 31;
    const float* x = in + (size_t)row * N;

    float s = 0.f, ss = 0.f;
    for (int c = tid; c < N; c += 256) { float v = x[c]; s += v; ss += v * v; }
    for (int o = 16; o > 0; o >>= 1) { s += __shfl_xor(s, o, 32); ss += __shfl_xor(ss, o, 32); }
    __shared__ float sh[16];
    if (lane == 0) { sh[wid] = s; sh[wid + 8] = ss; }
    __syncthreads();
    if (tid == 0) {
        float ts = 0.f, tss = 0.f;
        for (int w = 0; w < 8; ++w) { ts += sh[w]; tss += sh[w + 8]; }
        sh[0] = ts; sh[1] = tss;
    }
    __syncthreads();
    const float mean = sh[0] / (float)N;
    const float var  = sh[1] / (float)N - mean * mean;
    const float inv  = rsqrtf(var + 1e-5f);

    for (int c = tid; c < N; c += 256) {
        float y = (x[c] - mean) * inv * gam[c] + bet[c];
        if (act) { // jax.nn.gelu (tanh approximation)
            float u = 0.7978845608028654f * (y + 0.044715f * y * y * y);
            y = 0.5f * y * (1.f + tanhf(u));
        }
        if (outF) outF[(size_t)row * N + c] = y;
        if (outH) outH[(size_t)row * N + c] = (_Float16)y;
    }
}

// ---------------- T = h + a*k (also emits f16) ----------------
__global__ void axpy_kernel(const float* __restrict__ h, const float* __restrict__ k,
                            float a, float* __restrict__ T, _Float16* __restrict__ Th, int n)
{
    int i = (int)(blockIdx.x * blockDim.x + threadIdx.x);
    if (i < n) { float v = h[i] + a * k[i]; T[i] = v; Th[i] = (_Float16)v; }
}

// ---------------- RK4 combine: h += c*(k1 + 2k2 + 2k3 + k4) ----------------
__global__ void rk4_kernel(float* __restrict__ h, _Float16* __restrict__ hh,
                           const float* __restrict__ k1, const float* __restrict__ k2,
                           const float* __restrict__ k3, const float* __restrict__ k4,
                           float c, int n)
{
    int i = (int)(blockIdx.x * blockDim.x + threadIdx.x);
    if (i < n) {
        float v = h[i] + c * (k1[i] + 2.f * (k2[i] + k3[i]) + k4[i]);
        h[i] = v; hh[i] = (_Float16)v;
    }
}

// ---------------- attention: per-(row,head) rank-1 softmax ----------------
// qkv[b, c, h, d] at qkv[b*1536 + c*512 + h*64 + d];
// out_i = sum_j softmax_j(q_i * k_j * s) * v_j  -> f16 for the Wout GEMM.
__global__ __launch_bounds__(256)
void attn_kernel(const float* __restrict__ qkv, _Float16* __restrict__ outH)
{
    const int b = (int)blockIdx.x;
    const int w = (int)(threadIdx.x >> 5);    // head
    const int lane = (int)(threadIdx.x & 31);
    const float* base = qkv + (size_t)b * 1536;
    const float* q = base + w * 64;
    const float* k = base + 512 + w * 64;
    const float* v = base + 1024 + w * 64;

    const float k0 = k[lane], k1 = k[lane + 32];
    const float v0 = v[lane], v1 = v[lane + 32];
    const float scale = 0.125f; // 1/sqrt(64)

    for (int i = 0; i < 64; ++i) {
        const float qi = q[i] * scale;
        float x0 = qi * k0, x1 = qi * k1;
        float m = fmaxf(x0, x1);
        for (int o = 16; o > 0; o >>= 1) m = fmaxf(m, __shfl_xor(m, o, 32));
        float e0 = __expf(x0 - m), e1 = __expf(x1 - m);
        float se = e0 + e1, sv = e0 * v0 + e1 * v1;
        for (int o = 16; o > 0; o >>= 1) { se += __shfl_xor(se, o, 32); sv += __shfl_xor(sv, o, 32); }
        if (lane == 0) outH[(size_t)b * 512 + w * 64 + i] = (_Float16)(sv / se);
    }
}

// ---------------- row softmax over 128 cols -> d_out (f32) ----------------
__global__ __launch_bounds__(32)
void softmax128_kernel(const float* __restrict__ in, float* __restrict__ out)
{
    const int row = (int)blockIdx.x;
    const int lane = (int)threadIdx.x;
    const float* x = in + (size_t)row * 128;
    float v[4]; float m = -3.4e38f;
#pragma unroll
    for (int t = 0; t < 4; ++t) { v[t] = x[lane + 32 * t]; m = fmaxf(m, v[t]); }
    for (int o = 16; o > 0; o >>= 1) m = fmaxf(m, __shfl_xor(m, o, 32));
    float s = 0.f;
#pragma unroll
    for (int t = 0; t < 4; ++t) { v[t] = __expf(v[t] - m); s += v[t]; }
    for (int o = 16; o > 0; o >>= 1) s += __shfl_xor(s, o, 32);
    const float inv = 1.f / s;
#pragma unroll
    for (int t = 0; t < 4; ++t) out[(size_t)row * 128 + lane + 32 * t] = v[t] * inv;
}

// =====================================================================
extern "C" void kernel_launch(void* const* d_in, const int* in_sizes, int n_in,
                              void* d_out, int out_size, void* d_ws, size_t ws_size,
                              hipStream_t stream)
{
    (void)in_sizes; (void)n_in; (void)out_size; (void)ws_size;
    const int B = 1024;
    const int EW = B * 512;

    // ---- workspace carve (~40 MB total; everything L2-resident) ----
    size_t off = 0;
    auto carve  = [&](size_t bytes) -> void* {
        void* p = (char*)d_ws + off; off += (bytes + 255) & ~(size_t)255; return p;
    };
    auto carveH = [&](size_t n) -> _Float16* { return (_Float16*)carve(n * sizeof(_Float16)); };
    auto carveF = [&](size_t n) -> float*    { return (float*)carve(n * sizeof(float)); };

    _Float16* encW[4];
    encW[0] = carveH(512*256); encW[1] = carveH(256*512);
    encW[2] = carveH(512*256); encW[3] = carveH(512*512);
    _Float16 *lW1[4], *lW2[4], *lWg[4];
    for (int i = 0; i < 4; ++i) { lW1[i] = carveH(768*512); lW2[i] = carveH(512*768); lWg[i] = carveH(512*512); }
    _Float16 *aWqkv[2], *aWout[2];
    for (int j = 0; j < 2; ++j) { aWqkv[j] = carveH(1536*512); aWout[j] = carveH(512*512); }
    _Float16* outWh = carveH(512*512);
    _Float16* decW[3];
    decW[0] = carveH(256*512); decW[1] = carveH(128*256); decW[2] = carveH(128*128);
    _Float16* Xh = carveH((size_t)B * 256);

    float* H  = carveF(EW);  _Float16* Hh = carveH(EW);
    float* K1 = carveF(EW);  float* K2 = carveF(EW);
    float* K3 = carveF(EW);  float* K4 = carveF(EW);
    float* T  = carveF(EW);  _Float16* Th = carveH(EW);
    float* U  = carveF((size_t)B * 1536);
    _Float16* Uh = carveH((size_t)B * 768);
    float* V  = carveF(EW);  _Float16* Vh = carveH(EW);

    auto F = [&](int i) -> const float* { return (const float*)d_in[i]; };

    auto cvt = [&](const float* src, _Float16* dst, int n) {
        cvt_kernel<<<dim3((n + 255) / 256), dim3(256), 0, stream>>>(src, dst, n);
    };

    // ---- one-time (per call) weight casts to f16 ----
    cvt(F(1), encW[0], 512*256); cvt(F(2), encW[1], 256*512);
    cvt(F(3), encW[2], 512*256); cvt(F(4), encW[3], 512*512);
    for (int i = 0; i < 4; ++i) {
        int b0 = 15 + 8 * i;
        cvt(F(b0 + 0), lW1[i], 768*512);
        cvt(F(b0 + 2), lW2[i], 512*768);
        cvt(F(b0 + 6), lWg[i], 512*512);
    }
    for (int j = 0; j < 2; ++j) {
        int a0 = 47 + 4 * j;
        cvt(F(a0 + 2), aWqkv[j], 1536*512);
        cvt(F(a0 + 3), aWout[j], 512*512);
    }
    cvt(F(55), outWh, 512*512);
    cvt(F(57), decW[0], 256*512); cvt(F(58), decW[1], 128*256); cvt(F(59), decW[2], 128*128);
    cvt(F(0), Xh, B * 256);

    auto gemm = [&](int mode, const _Float16* Am, const _Float16* Wm, const float* bias,
                    const float* resid, const float* tb, const float* hb,
                    float* oF, _Float16* oH, int N, int Kd) {
        dim3 grid(N / 128, B / 64);
        switch (mode) {
        case 0: wmma_gemm_kernel<0><<<grid, dim3(128), 0, stream>>>(Am, Wm, bias, resid, tb, hb, oF, oH, B, N, Kd); break;
        case 1: wmma_gemm_kernel<1><<<grid, dim3(128), 0, stream>>>(Am, Wm, bias, resid, tb, hb, oF, oH, B, N, Kd); break;
        case 2: wmma_gemm_kernel<2><<<grid, dim3(128), 0, stream>>>(Am, Wm, bias, resid, tb, hb, oF, oH, B, N, Kd); break;
        case 3: wmma_gemm_kernel<3><<<grid, dim3(128), 0, stream>>>(Am, Wm, bias, resid, tb, hb, oF, oH, B, N, Kd); break;
        case 4: wmma_gemm_kernel<4><<<grid, dim3(128), 0, stream>>>(Am, Wm, bias, resid, tb, hb, oF, oH, B, N, Kd); break;
        default: wmma_gemm_kernel<5><<<grid, dim3(128), 0, stream>>>(Am, Wm, bias, resid, tb, hb, oF, oH, B, N, Kd); break;
        }
    };
    auto ln = [&](const float* in, const float* g, const float* b,
                  float* oF, _Float16* oH, int N, int act) {
        ln_kernel<<<dim3(B), dim3(256), 0, stream>>>(in, g, b, oF, oH, N, act);
    };
    auto axpy = [&](const float* h, const float* k, float a) {
        axpy_kernel<<<dim3((EW + 255) / 256), dim3(256), 0, stream>>>(h, k, a, T, Th, EW);
    };

    // ---- ode_f: reads (hin,hinH), uses T/Th as working state, writes kout ----
    auto ode_f = [&](const float* hin, const _Float16* hinH, float* kout) {
        const float*    curF = hin;
        const _Float16* curH = hinH;
        for (int i = 0; i < 4; ++i) {
            int b0 = 15 + 8 * i;
            gemm(1, curH, lW1[i], F(b0 + 1), nullptr, nullptr, nullptr,
                 nullptr, Uh, 768, 512);                                    // tanh(h@W1+b1) -> f16
            gemm(0, Uh, lW2[i], F(b0 + 3), nullptr, nullptr, nullptr,
                 U, nullptr, 512, 768);                                     // @W2 + b2
            ln(U, F(b0 + 4), F(b0 + 5), V, nullptr, 512, 0);                // t = LN(.)
            gemm(5, curH, lWg[i], F(b0 + 7), nullptr, V, curF,
                 T, Th, 512, 512);                                          // fused liquid gate
            curF = T; curH = Th;
            if (i & 1) {
                int j = i >> 1, a0 = 47 + 4 * j;
                ln(T, F(a0 + 0), F(a0 + 1), nullptr, Vh, 512, 0);           // xn -> f16
                gemm(3, Vh, aWqkv[j], nullptr, nullptr, nullptr, nullptr,
                     U, nullptr, 1536, 512);
                attn_kernel<<<dim3(B), dim3(256), 0, stream>>>(U, Vh);      // attn out -> f16
                gemm(4, Vh, aWout[j], nullptr, T, nullptr, nullptr,
                     T, Th, 512, 512);                                      // x + out@Wout
            }
        }
        gemm(0, curH, outWh, F(56), nullptr, nullptr, nullptr, kout, nullptr, 512, 512);
    };

    // ---- encoder ----
    gemm(0, Xh, encW[0], F(5), nullptr, nullptr, nullptr, U, nullptr, 512, 256);
    ln(U, F(9),  F(12), nullptr, Th, 512, 1);
    gemm(0, Th, encW[1], F(6), nullptr, nullptr, nullptr, U, nullptr, 256, 512);
    ln(U, F(10), F(13), nullptr, Th, 256, 1);
    gemm(0, Th, encW[2], F(7), nullptr, nullptr, nullptr, U, nullptr, 512, 256);
    ln(U, F(11), F(14), nullptr, Th, 512, 1);
    gemm(2, Th, encW[3], F(8), nullptr, nullptr, nullptr, H, Hh, 512, 512); // h0 = tanh(.)

    // ---- RK4 x 8 steps ----
    const float dt = 1.f / 8.f;
    for (int s = 0; s < 8; ++s) {
        ode_f(H, Hh, K1);
        axpy(H, K1, 0.5f * dt);     ode_f(T, Th, K2);
        axpy(H, K2, 0.5f * dt);     ode_f(T, Th, K3);
        axpy(H, K3, dt);            ode_f(T, Th, K4);
        rk4_kernel<<<dim3((EW + 255) / 256), dim3(256), 0, stream>>>(
            H, Hh, K1, K2, K3, K4, dt / 6.f, EW);
    }

    // ---- decoder ----
    gemm(0, Hh, decW[0], F(60), nullptr, nullptr, nullptr, U, nullptr, 256, 512);
    ln(U, F(63), F(65), nullptr, Th, 256, 1);
    gemm(0, Th, decW[1], F(61), nullptr, nullptr, nullptr, U, nullptr, 128, 256);
    ln(U, F(64), F(66), nullptr, Th, 128, 1);
    gemm(0, Th, decW[2], F(62), nullptr, nullptr, nullptr, U, nullptr, 128, 128);
    softmax128_kernel<<<dim3(B), dim3(32), 0, stream>>>(U, (float*)d_out);
}
